// QKVMergedCommonAttention_74612171866419
// MI455X (gfx1250) — matrix operand order repaired and checked by reading, not compiled, not run on hardware
//
#include <hip/hip_runtime.h>

// ---------------- problem constants ----------------
#define B_DIM   2
#define S_DIM   2048
#define HID     4096
#define NH      32
#define NKV     8
#define HD      128
#define GROUPS  (NH / NKV)          // 4
#define Q_SIZE  (NH * HD)           // 4096
#define KV_SIZE (NKV * HD)          // 1024
#define QKV_O   (Q_SIZE + 2 * KV_SIZE) // 6144
#define SCALING 0.08838834764831845f   // 128^-0.5

typedef __attribute__((ext_vector_type(16))) __bf16 v16bf;
typedef __attribute__((ext_vector_type(8)))  __bf16 v8bf;
typedef __attribute__((ext_vector_type(8)))  float  v8f;

#define HAVE_TDM __has_builtin(__builtin_amdgcn_tensor_load_to_lds)

// ---------------- bf16 helpers (bit-exact, no bf16 arithmetic) ----------------
__device__ __forceinline__ __bf16 f32_to_bf16(float f) {
    unsigned u = __builtin_bit_cast(unsigned, f);
    u += 0x7FFFu + ((u >> 16) & 1u);              // round-to-nearest-even
    unsigned short h = (unsigned short)(u >> 16);
    return __builtin_bit_cast(__bf16, h);
}
__device__ __forceinline__ float bf16_to_f32(__bf16 b) {
    unsigned short h = __builtin_bit_cast(unsigned short, b);
    unsigned u = ((unsigned)h) << 16;
    return __builtin_bit_cast(float, u);
}

// Build a 16-element bf16 fragment from two 8-element (16B) chunks.
__device__ __forceinline__ v16bf load_frag16(const __bf16* p, int o0, int o1) {
    v8bf lo = *(const v8bf*)(p + o0);
    v8bf hi = *(const v8bf*)(p + o1);
    v16bf r;
#pragma unroll
    for (int i = 0; i < 8; ++i) { r[i] = lo[i]; r[i + 8] = hi[i]; }
    return r;
}

__device__ __forceinline__ void store_out(float* p, float v)  { *p = v; }
__device__ __forceinline__ void store_out(__bf16* p, float v) { *p = f32_to_bf16(v); }

// ---------------- TDM 2-D tile load: global (strided) -> LDS (packed) ----------------
#if HAVE_TDM
typedef __attribute__((ext_vector_type(4))) unsigned tdm_v4u;
typedef __attribute__((ext_vector_type(8))) int      tdm_v8i;
typedef __attribute__((ext_vector_type(4))) int      tdm_v4i;

// Packs D# group0/group1 per CDNA5 ISA.
//   data_size=1 (2-byte elems), 2-D tile (tile_dim2=0, groups 2/3 zero),
//   tile rows stored contiguously in LDS (no pad), no multicast / iterate.
__device__ __forceinline__ void tdm_load_tile_2d(unsigned lds_off, const void* gptr,
                                                 unsigned tile_w, unsigned tile_h,
                                                 unsigned tensor_w, unsigned tensor_h,
                                                 unsigned row_stride /* elems */)
{
    unsigned long long ga = (unsigned long long)(uintptr_t)gptr;
    tdm_v4u g0;
    g0[0] = 1u;                                       // count=1, user descriptor
    g0[1] = lds_off;                                  // lds_addr [63:32]
    g0[2] = (unsigned)(ga & 0xFFFFFFFFu);             // global_addr [95:64]
    g0[3] = (unsigned)((ga >> 32) & 0x01FFFFFFu)      // global_addr [120:96]
          | (2u << 30);                               // type=2 ("image")
    tdm_v8i g1;
    g1[0] = (int)(1u << 16);                          // workgroup_mask=0, data_size=1 (2B)
    g1[1] = (int)((tensor_w & 0xFFFFu) << 16);        // tensor_dim0 [79:48] lo
    g1[2] = (int)(((tensor_w >> 16) & 0xFFFFu)        // tensor_dim0 hi
          | ((tensor_h & 0xFFFFu) << 16));            // tensor_dim1 [111:80] lo
    g1[3] = (int)(((tensor_h >> 16) & 0xFFFFu)        // tensor_dim1 hi
          | ((tile_w & 0xFFFFu) << 16));              // tile_dim0 [127:112]
    g1[4] = (int)(tile_h & 0xFFFFu);                  // tile_dim1; tile_dim2=0
    g1[5] = (int)row_stride;                          // tensor_dim0_stride [207:160] lo
    g1[6] = 0;                                        // stride hi, dim1_stride lo
    g1[7] = 0;                                        // dim1_stride hi
    tdm_v4i gz = {0, 0, 0, 0};
#if __clang_major__ >= 23
    tdm_v8i gz8 = {0, 0, 0, 0, 0, 0, 0, 0};
    __builtin_amdgcn_tensor_load_to_lds(g0, g1, gz, gz, gz8, 0);
#else
    __builtin_amdgcn_tensor_load_to_lds(g0, g1, gz, gz, 0);
#endif
}
#endif // HAVE_TDM

// ---------------- fp32 -> bf16 convert ----------------
__global__ __launch_bounds__(256)
void f32_to_bf16_kernel(const float* __restrict__ in, __bf16* __restrict__ out, size_t n) {
    size_t i = (size_t)blockIdx.x * blockDim.x + threadIdx.x;
    if (i < n) out[i] = f32_to_bf16(in[i]);
}

// ---------------- WMMA GEMM: C[M,N] = A[M,K] * B[N,K]^T ----------------
// Block = 256 threads = 8 waves (4 along M x 2 along N). Block tile 128x128,
// wave tile 32x64 (2x4 frags of 16x16 f32 accum). K step = 32 (bf16 WMMA K).
// Tile staging: TDM tensor_load_to_lds (TENSORcnt) when available; else
// cooperative global_load_b128 -> ds_store.
template <typename OutT>
__global__ __launch_bounds__(256)
void gemm_bf16_wmma(const __bf16* __restrict__ A, const __bf16* __restrict__ Bm,
                    OutT* __restrict__ C, int M, int N, int K)
{
#if HAVE_TDM
    __shared__ __align__(16) __bf16 sA[128][32];   // packed: TDM writes rows contiguously
    __shared__ __align__(16) __bf16 sB[128][32];
    const int LDS_STRIDE = 32;
#else
    __shared__ __align__(16) __bf16 sA[128][40];   // padded for ds_store path
    __shared__ __align__(16) __bf16 sB[128][40];
    const int LDS_STRIDE = 40;
#endif

    const int t    = threadIdx.x;
    const int lane = t & 31;
    const int wave = t >> 5;
    const int half = lane >> 4;   // 0: lanes 0-15, 1: lanes 16-31
    const int l15  = lane & 15;
    const int wm   = wave & 3;    // wave row in block (4)
    const int wn   = wave >> 2;   // wave col in block (2)
    const int m0   = blockIdx.y * 128;
    const int n0   = blockIdx.x * 128;

    v8f acc[2][4] = {};

    for (int kt = 0; kt < K; kt += 32) {
#if HAVE_TDM
        if (wave == 0) {
            // DMA both 128x32 bf16 tiles straight into LDS, then drain TENSORcnt.
            tdm_load_tile_2d((unsigned)(uintptr_t)&sA[0][0],
                             &A[(size_t)m0 * K + kt], 32u, 128u,
                             (unsigned)K, (unsigned)M, (unsigned)K);
            tdm_load_tile_2d((unsigned)(uintptr_t)&sB[0][0],
                             &Bm[(size_t)n0 * K + kt], 32u, 128u,
                             (unsigned)K, (unsigned)N, (unsigned)K);
            __builtin_amdgcn_s_wait_tensorcnt(0);
        }
        if (kt + 64 <= K) {               // global_prefetch_b8 of next tiles
            __builtin_prefetch(&A[(size_t)(m0 + (t >> 1)) * K + kt + 32], 0, 1);
            __builtin_prefetch(&Bm[(size_t)(n0 + (t >> 1)) * K + kt + 32], 0, 1);
        }
        __syncthreads();
#else
        // cooperative stage of A (128x32) and B (128x32) tiles, 16B per access
#pragma unroll
        for (int j = 0; j < 2; ++j) {
            int seg = t * 2 + j;          // 0..511
            int row = seg >> 2;
            int col = (seg & 3) * 8;
            *(v8bf*)&sA[row][col] = *(const v8bf*)&A[(size_t)(m0 + row) * K + kt + col];
            *(v8bf*)&sB[row][col] = *(const v8bf*)&Bm[(size_t)(n0 + row) * K + kt + col];
        }
        if (kt + 64 <= K) {
            __builtin_prefetch(&A[(size_t)(m0 + (t >> 1)) * K + kt + 32], 0, 1);
            __builtin_prefetch(&Bm[(size_t)(n0 + (t >> 1)) * K + kt + 32], 0, 1);
        }
        __syncthreads();
#endif

        // A frags: lane l15 = row, halves split K runs {0-7,16-23}/{8-15,24-31}
        v16bf af[2];
#pragma unroll
        for (int mf = 0; mf < 2; ++mf) {
            const __bf16* ap = &sA[wm * 32 + mf * 16 + l15][0];
            (void)LDS_STRIDE;
            af[mf] = load_frag16(ap, half * 8, 16 + half * 8);
        }
        // B frags: lane l15 = column, halves split K runs {0-7,16-23}/{8-15,24-31}
#pragma unroll
        for (int nf = 0; nf < 4; ++nf) {
            const __bf16* bp = &sB[wn * 64 + nf * 16 + l15][0];
            v16bf bf = load_frag16(bp, half * 8, 16 + half * 8);
#pragma unroll
            for (int mf = 0; mf < 2; ++mf) {
                acc[mf][nf] = __builtin_amdgcn_wmma_f32_16x16x32_bf16(
                    false, af[mf], false, bf, (short)0, acc[mf][nf], false, false);
            }
        }
        __syncthreads();
    }

    // C layout: vgpr r -> M = r + 8*half, N = l15
#pragma unroll
    for (int mf = 0; mf < 2; ++mf)
#pragma unroll
        for (int nf = 0; nf < 4; ++nf)
#pragma unroll
            for (int r = 0; r < 8; ++r) {
                int row = m0 + wm * 32 + mf * 16 + r + half * 8;
                int col = n0 + wn * 64 + nf * 16 + l15;
                store_out(&C[(size_t)row * N + col], acc[mf][nf][r]);
            }
}

// ---------------- RoPE + head split (q,k roped; v copied) ----------------
__global__ __launch_bounds__(256)
void rope_split_kernel(const __bf16* __restrict__ qkv,
                       const float* __restrict__ cosb, const float* __restrict__ sinb,
                       __bf16* __restrict__ Qo, __bf16* __restrict__ Ko, __bf16* __restrict__ Vo)
{
    const size_t NQ  = (size_t)B_DIM * S_DIM * NH  * HD;
    const size_t NKE = (size_t)B_DIM * S_DIM * NKV * HD;
    size_t idx = (size_t)blockIdx.x * blockDim.x + threadIdx.x;

    if (idx < NQ) {
        int d  = (int)(idx % HD);
        int hh = (int)((idx / HD) % NH);
        int s  = (int)((idx / ((size_t)HD * NH)) % S_DIM);
        int b  = (int)(idx / ((size_t)HD * NH * S_DIM));
        size_t row = (size_t)b * S_DIM + s;
        const __bf16* src = qkv + row * QKV_O + hh * HD;
        float x  = bf16_to_f32(src[d]);
        float x2 = bf16_to_f32(src[(d < 64) ? d + 64 : d - 64]);
        float c  = cosb[row * HD + d];
        float sn = sinb[row * HD + d];
        float o  = x * c + ((d < 64) ? -x2 : x2) * sn;
        Qo[(((size_t)b * NH + hh) * S_DIM + s) * HD + d] = f32_to_bf16(o);
    } else if (idx < NQ + NKE) {
        size_t i = idx - NQ;
        int d   = (int)(i % HD);
        int kvh = (int)((i / HD) % NKV);
        int s   = (int)((i / ((size_t)HD * NKV)) % S_DIM);
        int b   = (int)(i / ((size_t)HD * NKV * S_DIM));
        size_t row = (size_t)b * S_DIM + s;
        const __bf16* src = qkv + row * QKV_O + Q_SIZE + kvh * HD;
        float x  = bf16_to_f32(src[d]);
        float x2 = bf16_to_f32(src[(d < 64) ? d + 64 : d - 64]);
        float c  = cosb[row * HD + d];
        float sn = sinb[row * HD + d];
        float o  = x * c + ((d < 64) ? -x2 : x2) * sn;
        Ko[(((size_t)b * NKV + kvh) * S_DIM + s) * HD + d] = f32_to_bf16(o);
    } else if (idx < NQ + 2 * NKE) {
        size_t i = idx - NQ - NKE;
        int d   = (int)(i % HD);
        int kvh = (int)((i / HD) % NKV);
        int s   = (int)((i / ((size_t)HD * NKV)) % S_DIM);
        int b   = (int)(i / ((size_t)HD * NKV * S_DIM));
        size_t row = (size_t)b * S_DIM + s;
        Vo[(((size_t)b * NKV + kvh) * S_DIM + s) * HD + d] =
            qkv[row * QKV_O + Q_SIZE + KV_SIZE + kvh * HD + d];
    }
}

// ---------------- WMMA flash attention (causal, GQA) ----------------
// Block = 128 threads = 4 waves; block owns 64 query rows of one (b,h);
// wave owns a 16-row strip. 64-key blocks streamed with online softmax.
__global__ __launch_bounds__(128)
void flash_attn_wmma(const __bf16* __restrict__ Q, const __bf16* __restrict__ Kc,
                     const __bf16* __restrict__ Vc, __bf16* __restrict__ Out)
{
    __shared__ __align__(16) __bf16 ldsVt[HD][72];      // V^T [d][key] for 64-key block
    __shared__ __align__(16) __bf16 ldsP[4][16][72];    // P restage per wave (C->A layout)

    const int t    = threadIdx.x;
    const int lane = t & 31;
    const int wave = t >> 5;
    const int half = lane >> 4;
    const int l15  = lane & 15;
    const int qb   = blockIdx.x;          // 64-row query block
    const int bh   = blockIdx.y;
    const int b    = bh / NH, h = bh % NH;
    const int kvh  = h / GROUPS;
    const int q0   = qb * 64 + wave * 16; // wave's first query row

    const __bf16* Qbase = Q  + (size_t)bh * S_DIM * HD;
    const __bf16* Kbase = Kc + (size_t)(b * NKV + kvh) * S_DIM * HD;
    const __bf16* Vbase = Vc + (size_t)(b * NKV + kvh) * S_DIM * HD;

    // Q fragments held in registers for the whole pass (16 x 128 per wave)
    v16bf qa[4];
    {
        const __bf16* qrow = Qbase + (size_t)(q0 + l15) * HD;
#pragma unroll
        for (int kc = 0; kc < 4; ++kc)
            qa[kc] = load_frag16(qrow, kc * 32 + half * 8, kc * 32 + 16 + half * 8);
    }

    float m_r[8], l_r[8];
    v8f o_acc[8];
    v8f vzero = {};
#pragma unroll
    for (int r = 0; r < 8; ++r) { m_r[r] = -3.0e38f; l_r[r] = 0.f; }
#pragma unroll
    for (int df = 0; df < 8; ++df) o_acc[df] = vzero;

    for (int kb = 0; kb <= qb; ++kb) {
        // ---- stage V^T into LDS (2 threads per key row) ----
        {
            int key   = t >> 1;
            int dbase = (t & 1) * 64;
            const __bf16* vrow = Vbase + (size_t)(kb * 64 + key) * HD + dbase;
#pragma unroll
            for (int j = 0; j < 8; ++j) {
                v8bf chunk = *(const v8bf*)(vrow + j * 8);
#pragma unroll
                for (int e = 0; e < 8; ++e)
                    ldsVt[dbase + j * 8 + e][key] = chunk[e];
            }
        }
        __syncthreads();

        // ---- S = Q K^T (K frags straight from global; K cache is [key][d]) ----
        v8f sf[4];
#pragma unroll
        for (int nf = 0; nf < 4; ++nf) {
            v8f s = vzero;
            const __bf16* krow = Kbase + (size_t)(kb * 64 + nf * 16 + l15) * HD;
#pragma unroll
            for (int kc = 0; kc < 4; ++kc) {
                v16bf bfr = load_frag16(krow, kc * 32 + half * 8, kc * 32 + 16 + half * 8);
                s = __builtin_amdgcn_wmma_f32_16x16x32_bf16(
                        false, qa[kc], false, bfr, (short)0, s, false, false);
            }
            sf[nf] = s;
        }

        // ---- scale + causal mask ----
#pragma unroll
        for (int nf = 0; nf < 4; ++nf) {
            int key = kb * 64 + nf * 16 + l15;
#pragma unroll
            for (int r = 0; r < 8; ++r) {
                int qrow = q0 + r + half * 8;
                float sv = sf[nf][r] * SCALING;
                sf[nf][r] = (key > qrow) ? -3.0e38f : sv;
            }
        }

        // ---- online softmax (row reductions across 16-lane groups) ----
        float pv[4][8];
#pragma unroll
        for (int r = 0; r < 8; ++r) {
            float mx = -3.0e38f;
#pragma unroll
            for (int nf = 0; nf < 4; ++nf) mx = fmaxf(mx, sf[nf][r]);
#pragma unroll
            for (int off = 8; off >= 1; off >>= 1)
                mx = fmaxf(mx, __shfl_xor(mx, off, 16));
            float mnew = fmaxf(m_r[r], mx);
            float corr = __expf(m_r[r] - mnew);
            float rs = 0.f;
#pragma unroll
            for (int nf = 0; nf < 4; ++nf) {
                float p = __expf(sf[nf][r] - mnew);
                pv[nf][r] = p; rs += p;
            }
#pragma unroll
            for (int off = 8; off >= 1; off >>= 1)
                rs += __shfl_xor(rs, off, 16);
            m_r[r] = mnew;
            l_r[r] = l_r[r] * corr + rs;
#pragma unroll
            for (int df = 0; df < 8; ++df) o_acc[df][r] *= corr;
        }

        // ---- restage P (C layout -> A layout) through LDS as bf16 ----
#pragma unroll
        for (int nf = 0; nf < 4; ++nf)
#pragma unroll
            for (int r = 0; r < 8; ++r)
                ldsP[wave][r + half * 8][nf * 16 + l15] = f32_to_bf16(pv[nf][r]);
        __syncthreads();

        // ---- O += P V ----
        v16bf pa[2];
        {
            const __bf16* prow = &ldsP[wave][l15][0];
#pragma unroll
            for (int kc = 0; kc < 2; ++kc)
                pa[kc] = load_frag16(prow, kc * 32 + half * 8, kc * 32 + 16 + half * 8);
        }
#pragma unroll
        for (int df = 0; df < 8; ++df) {
            const __bf16* vtp = &ldsVt[df * 16 + l15][0];
#pragma unroll
            for (int kc = 0; kc < 2; ++kc) {
                v16bf vfr = load_frag16(vtp, kc * 32 + half * 8, kc * 32 + 16 + half * 8);
                o_acc[df] = __builtin_amdgcn_wmma_f32_16x16x32_bf16(
                                false, pa[kc], false, vfr, (short)0, o_acc[df], false, false);
            }
        }
        __syncthreads();   // protect ldsVt/ldsP for next key block
    }

    // ---- epilogue: normalize, write [b][s][h*HD+d] as bf16 ----
#pragma unroll
    for (int r = 0; r < 8; ++r) {
        int qrow = q0 + r + half * 8;
        float inv = 1.0f / l_r[r];
#pragma unroll
        for (int df = 0; df < 8; ++df) {
            int d = df * 16 + l15;
            Out[((size_t)(b * S_DIM + qrow) * NH + h) * HD + d] =
                f32_to_bf16(o_acc[df][r] * inv);
        }
    }
}

// ---------------- launch ----------------
extern "C" void kernel_launch(void* const* d_in, const int* in_sizes, int n_in,
                              void* d_out, int out_size, void* d_ws, size_t ws_size,
                              hipStream_t stream)
{
    const float* hs   = (const float*)d_in[0];
    const float* cosb = (const float*)d_in[1];
    const float* sinb = (const float*)d_in[2];
    const float* wqkv = (const float*)d_in[3];
    const float* wo   = (const float*)d_in[4];
    float* out = (float*)d_out;

    const size_t N_HS   = (size_t)B_DIM * S_DIM * HID;     // 16,777,216
    const size_t N_WQKV = (size_t)QKV_O * HID;             // 25,165,824
    const size_t N_WO   = (size_t)HID * Q_SIZE;            // 16,777,216
    const size_t N_QKV  = (size_t)B_DIM * S_DIM * QKV_O;   // 25,165,824
    const size_t N_Q    = (size_t)B_DIM * NH  * S_DIM * HD;
    const size_t N_KV   = (size_t)B_DIM * NKV * S_DIM * HD;
    const size_t N_AO   = (size_t)B_DIM * S_DIM * Q_SIZE;

    char* ws = (char*)d_ws;
    __bf16* hs_b   = (__bf16*)ws; ws += N_HS   * 2;
    __bf16* wqkv_b = (__bf16*)ws; ws += N_WQKV * 2;
    __bf16* wo_b   = (__bf16*)ws; ws += N_WO   * 2;
    __bf16* qkv_b  = (__bf16*)ws; ws += N_QKV  * 2;
    __bf16* q_b    = (__bf16*)ws; ws += N_Q    * 2;
    __bf16* k_b    = (__bf16*)ws; ws += N_KV   * 2;
    __bf16* v_b    = (__bf16*)ws; ws += N_KV   * 2;
    __bf16* ao_b   = (__bf16*)ws; ws += N_AO   * 2;

    // 1) fp32 -> bf16 conversions
    f32_to_bf16_kernel<<<(unsigned)((N_HS   + 255) / 256), 256, 0, stream>>>(hs,   hs_b,   N_HS);
    f32_to_bf16_kernel<<<(unsigned)((N_WQKV + 255) / 256), 256, 0, stream>>>(wqkv, wqkv_b, N_WQKV);
    f32_to_bf16_kernel<<<(unsigned)((N_WO   + 255) / 256), 256, 0, stream>>>(wo,   wo_b,   N_WO);

    // 2) qkv = hs @ w_qkv^T   (M=4096, N=6144, K=4096)
    gemm_bf16_wmma<__bf16><<<dim3(QKV_O / 128, (B_DIM * S_DIM) / 128), 256, 0, stream>>>(
        hs_b, wqkv_b, qkv_b, B_DIM * S_DIM, QKV_O, HID);

    // 3) RoPE + split into per-head layouts
    size_t nrope = N_Q + 2 * N_KV;
    rope_split_kernel<<<(unsigned)((nrope + 255) / 256), 256, 0, stream>>>(
        qkv_b, cosb, sinb, q_b, k_b, v_b);

    // 4) causal GQA flash attention
    flash_attn_wmma<<<dim3(S_DIM / 64, B_DIM * NH), 128, 0, stream>>>(q_b, k_b, v_b, ao_b);

    // 5) out = attn_out @ w_o^T  (M=4096, N=4096, K=4096), fp32 store
    gemm_bf16_wmma<float><<<dim3(HID / 128, (B_DIM * S_DIM) / 128), 256, 0, stream>>>(
        ao_b, wo_b, out, B_DIM * S_DIM, HID, HID);
}
